// CustomPokemonModel_53532472377491
// MI455X (gfx1250) — matrix operand order repaired and checked by reading, not compiled, not run on hardware
//
#include <hip/hip_runtime.h>
#include <hip/hip_bf16.h>
#include <math.h>

typedef __attribute__((ext_vector_type(16))) _Float16 v16h;
typedef __attribute__((ext_vector_type(8)))  float    v8f;
typedef int v4i_ __attribute__((vector_size(16)));

#define NB      4096      // batch
#define ED      128
#define LH      128
#define HD      64
#define IN_DIM  7808
#define KDIM    7936      // IN_DIM + LH (h0 fused)
#define NGATE   512       // 4*LH
#define KSTEP   64
#define NITER   (KDIM / KSTEP)   // 124

#if __has_builtin(__builtin_amdgcn_global_load_async_to_lds_b128)
#define HAVE_ASYNC_LDS 1
#else
#define HAVE_ASYNC_LDS 0
#endif

__device__ __forceinline__ void wait_async0() {
#if __has_builtin(__builtin_amdgcn_s_wait_asynccnt)
  __builtin_amdgcn_s_wait_asynccnt(0);
#else
  asm volatile("s_wait_asynccnt 0x0" ::: "memory");
#endif
}

// issue one 32-byte B-tile chunk: global (f16) -> LDS
__device__ __forceinline__ void issue_B(const _Float16* g, _Float16* l) {
#if HAVE_ASYNC_LDS
  __builtin_amdgcn_global_load_async_to_lds_b128(
      (__attribute__((address_space(1))) v4i_*)(g),
      (__attribute__((address_space(3))) v4i_*)(l), 0, 0);
  __builtin_amdgcn_global_load_async_to_lds_b128(
      (__attribute__((address_space(1))) v4i_*)(g + 8),
      (__attribute__((address_space(3))) v4i_*)(l + 8), 0, 0);
#else
  ((uint4*)l)[0] = ((const uint4*)g)[0];
  ((uint4*)l)[1] = ((const uint4*)g)[1];
#endif
}

// ---------------------------------------------------------------------------
// Prep 1: W_ext[512][7936] f16 = [Wih | Whh] row-major
// ---------------------------------------------------------------------------
__global__ void wext_kernel(const float* __restrict__ Wih,
                            const float* __restrict__ Whh,
                            _Float16* __restrict__ W_ext) {
  int idx = blockIdx.x * blockDim.x + threadIdx.x;   // exactly 512*KDIM threads
  int n = idx / KDIM;
  int k = idx - n * KDIM;
  float v = (k < IN_DIM) ? Wih[(size_t)n * IN_DIM + k]
                         : Whh[(size_t)n * LH + (k - IN_DIM)];
  W_ext[idx] = (_Float16)v;
}

// ---------------------------------------------------------------------------
// Prep 2: num_pre[B][128] f16 = numerical @ num_W.T + num_b
// ---------------------------------------------------------------------------
__global__ void num_proj_kernel(const float* __restrict__ numerical,
                                const float* __restrict__ num_W,
                                const float* __restrict__ num_b,
                                _Float16* __restrict__ num_pre) {
  int idx = blockIdx.x * blockDim.x + threadIdx.x;   // NB*128
  int b = idx >> 7, e = idx & 127;
  const float* nrow = numerical + (size_t)b * 84;
  const float* wrow = num_W + (size_t)e * 84;
  float s = num_b[e];
  #pragma unroll 4
  for (int k = 0; k < 84; ++k) s += nrow[k] * wrow[k];
  num_pre[idx] = (_Float16)s;
}

// ---------------------------------------------------------------------------
// Gather one 16-half A chunk (row gm, cols gk..gk+15 of x_ext), f16
// ---------------------------------------------------------------------------
__device__ __forceinline__ v16h gather16(const int*   __restrict__ ability_ids,
                                         const int*   __restrict__ move_ids,
                                         const float* __restrict__ ability_emb,
                                         const float* __restrict__ move_emb,
                                         const _Float16* __restrict__ num_pre,
                                         const float* __restrict__ h0,
                                         int gm, int gk) {
  v16h r;
  if (gk < 7680) {                       // embedding gathers (f32 -> f16)
    const float* src;
    if (gk < 1536) {
      int slot = gk >> 7;
      int id   = ability_ids[gm * 12 + slot];
      src = ability_emb + (size_t)id * ED + (gk & 127);
    } else {
      int k2   = gk - 1536;
      int slot = k2 >> 7;
      int id   = move_ids[gm * 48 + slot];
      src = move_emb + (size_t)id * ED + (k2 & 127);
    }
    const float4* s4 = (const float4*)src;
    float4 f0 = s4[0], f1 = s4[1], f2 = s4[2], f3 = s4[3];
    r[0]  = (_Float16)f0.x; r[1]  = (_Float16)f0.y;
    r[2]  = (_Float16)f0.z; r[3]  = (_Float16)f0.w;
    r[4]  = (_Float16)f1.x; r[5]  = (_Float16)f1.y;
    r[6]  = (_Float16)f1.z; r[7]  = (_Float16)f1.w;
    r[8]  = (_Float16)f2.x; r[9]  = (_Float16)f2.y;
    r[10] = (_Float16)f2.z; r[11] = (_Float16)f2.w;
    r[12] = (_Float16)f3.x; r[13] = (_Float16)f3.y;
    r[14] = (_Float16)f3.z; r[15] = (_Float16)f3.w;
  } else if (gk < IN_DIM) {              // numerical projection (already f16)
    r = *(const v16h*)(num_pre + (size_t)gm * ED + (gk - 7680));
  } else {                               // h0 columns (f32 -> f16)
    const float4* s4 = (const float4*)(h0 + (size_t)gm * LH + (gk - IN_DIM));
    float4 f0 = s4[0], f1 = s4[1], f2 = s4[2], f3 = s4[3];
    r[0]  = (_Float16)f0.x; r[1]  = (_Float16)f0.y;
    r[2]  = (_Float16)f0.z; r[3]  = (_Float16)f0.w;
    r[4]  = (_Float16)f1.x; r[5]  = (_Float16)f1.y;
    r[6]  = (_Float16)f1.z; r[7]  = (_Float16)f1.w;
    r[8]  = (_Float16)f2.x; r[9]  = (_Float16)f2.y;
    r[10] = (_Float16)f2.z; r[11] = (_Float16)f2.w;
    r[12] = (_Float16)f3.x; r[13] = (_Float16)f3.y;
    r[14] = (_Float16)f3.z; r[15] = (_Float16)f3.w;
  }
  return r;
}

// ---------------------------------------------------------------------------
// Main GEMM: gates[B][512] = x_ext @ W_ext^T  (gather fused into A-tile load)
// 256 threads (8 waves), tile 128(M) x 64(N), K-step 64, double-buffered LDS,
// async global->LDS for the B tile, software-pipelined A gather.
// ---------------------------------------------------------------------------
__global__ void __launch_bounds__(256)
gates_gemm_kernel(const int*   __restrict__ ability_ids,
                  const int*   __restrict__ move_ids,
                  const float* __restrict__ ability_emb,
                  const float* __restrict__ move_emb,
                  const _Float16* __restrict__ num_pre,
                  const float* __restrict__ h0,
                  const _Float16* __restrict__ W_ext,
                  float* __restrict__ gates) {
  __shared__ __align__(32) _Float16 As[2][128 * 64];   // [m][k]
  __shared__ __align__(32) _Float16 Bs[2][64 * 64];    // [n][k]

  const int m0 = (blockIdx.x >> 3) * 128;   // 32 M-tiles
  const int n0 = (blockIdx.x & 7) * 64;     // 8  N-tiles
  const int t    = threadIdx.x;
  const int lane = t & 31;
  const int wv   = t >> 5;                  // wave id 0..7

  // B-tile chunk owned by this thread (one 32B run per iteration)
  const int tn  = t >> 2;
  const int tkc = (t & 3) << 4;
  const _Float16* wrow = W_ext + (size_t)(n0 + tn) * KDIM + tkc;

  // A-tile chunks owned by this thread (two 32B runs per iteration)
  const int am0 = t >> 2,          akc0 = (t & 3) << 4;
  const int am1 = (t + 256) >> 2,  akc1 = tkc;   // (t+256)&3 == t&3

  v8f acc0 = {}, acc1 = {}, acc2 = {}, acc3 = {};

  // ---- prologue: fill buffer 0 with tile 0 ----
  issue_B(wrow, &Bs[0][tn * 64 + tkc]);
  v16h a0 = gather16(ability_ids, move_ids, ability_emb, move_emb, num_pre, h0,
                     m0 + am0, akc0);
  v16h a1 = gather16(ability_ids, move_ids, ability_emb, move_emb, num_pre, h0,
                     m0 + am1, akc1);
  *(v16h*)&As[0][am0 * 64 + akc0] = a0;
  *(v16h*)&As[0][am1 * 64 + akc1] = a1;
  wait_async0();
  __syncthreads();

  // fragment coordinates (constant across iterations)
  const int mrow = wv * 16 + (lane & 15);
  const int khi  = (lane >> 4) << 3;      // unused: A, like B, uses kb (K+16) for lanes 16-31
  const int nc   = lane & 15;
  const int kb   = (lane >> 4) << 4;      // B: lanes 16-31 hold K+16 halves

  int cur = 0;
  for (int it = 0; it < NITER; ++it) {
    const int nxt = cur ^ 1;
    const bool more = (it + 1) < NITER;
    const int knext = (it + 1) * KSTEP;

    // ---- issue next tile's loads before computing on current ----
    if (more) {
      issue_B(wrow + knext, &Bs[nxt][tn * 64 + tkc]);
      a0 = gather16(ability_ids, move_ids, ability_emb, move_emb, num_pre, h0,
                    m0 + am0, knext + akc0);
      a1 = gather16(ability_ids, move_ids, ability_emb, move_emb, num_pre, h0,
                    m0 + am1, knext + akc1);
    }

    // ---- compute on current buffers ----
    const _Float16* as = &As[cur][0];
    const _Float16* bs = &Bs[cur][0];
    #pragma unroll
    for (int kk = 0; kk < KSTEP; kk += 32) {
      // A fragment 16x32 f16, contiguous K: k = 16*(lane/16) + s, same as B (paired-K -> b32/b128 ds loads)
      v16h a;
      const _Float16* arow = as + mrow * 64 + kk;
      #pragma unroll
      for (int v = 0; v < 8; ++v) {
        int k0 = 2 * v + kb;
        a[2 * v]     = arow[k0];
        a[2 * v + 1] = arow[k0 + 1];
      }
      // B fragments 32x16 f16: lane = col N%16, contiguous 16 K-halves
      v16h b0 = *(const v16h*)(bs + (0  + nc) * 64 + kk + kb);
      v16h b1 = *(const v16h*)(bs + (16 + nc) * 64 + kk + kb);
      v16h b2 = *(const v16h*)(bs + (32 + nc) * 64 + kk + kb);
      v16h b3 = *(const v16h*)(bs + (48 + nc) * 64 + kk + kb);
      acc0 = __builtin_amdgcn_wmma_f32_16x16x32_f16(false, a, false, b0, (short)0, acc0, false, false);
      acc1 = __builtin_amdgcn_wmma_f32_16x16x32_f16(false, a, false, b1, (short)0, acc1, false, false);
      acc2 = __builtin_amdgcn_wmma_f32_16x16x32_f16(false, a, false, b2, (short)0, acc2, false, false);
      acc3 = __builtin_amdgcn_wmma_f32_16x16x32_f16(false, a, false, b3, (short)0, acc3, false, false);
    }

    // ---- commit next A tile to LDS, wait for async B, flip buffers ----
    if (more) {
      *(v16h*)&As[nxt][am0 * 64 + akc0] = a0;
      *(v16h*)&As[nxt][am1 * 64 + akc1] = a1;
    }
    wait_async0();
    __syncthreads();
    cur = nxt;
  }

  // ---- epilogue: C/D f32 layout: lane = N%16, VGPR v -> M = v + 8*(lane>=16)
  const int mr = m0 + wv * 16 + ((lane >> 4) << 3);
  #pragma unroll
  for (int v = 0; v < 8; ++v) {
    float* grow = gates + (size_t)(mr + v) * NGATE + n0 + nc;
    grow[0]  = acc0[v];
    grow[16] = acc1[v];
    grow[32] = acc2[v];
    grow[48] = acc3[v];
  }
}

// ---------------------------------------------------------------------------
// LSTM elementwise: gates (+bih+bhh) -> h1, c1  (written into d_out)
// ---------------------------------------------------------------------------
__global__ void lstm_ew_kernel(const float* __restrict__ gates,
                               const float* __restrict__ bih,
                               const float* __restrict__ bhh,
                               const float* __restrict__ c0,
                               float* __restrict__ out) {
  int idx = blockIdx.x * blockDim.x + threadIdx.x;   // NB*128
  int b = idx >> 7, j = idx & 127;
  const float* g = gates + (size_t)b * NGATE;
  float gi = g[j]       + bih[j]       + bhh[j];
  float gf = g[128 + j] + bih[128 + j] + bhh[128 + j];
  float gg = g[256 + j] + bih[256 + j] + bhh[256 + j];
  float go = g[384 + j] + bih[384 + j] + bhh[384 + j];
  float i_ = 1.0f / (1.0f + __expf(-gi));
  float f_ = 1.0f / (1.0f + __expf(-gf));
  float o_ = 1.0f / (1.0f + __expf(-go));
  float g_ = tanhf(gg);
  float c1 = f_ * c0[idx] + i_ * g_;
  float h1 = o_ * tanhf(c1);
  out[NB * 9 + idx]            = h1;  // h1 block
  out[NB * 9 + NB * 128 + idx] = c1;  // c1 block
}

// ---------------------------------------------------------------------------
// Head: relu(h1@W1.T+b1)@W2.T+b2 -> logits -> softmax -> masked renorm
// one 128-thread block per batch row
// ---------------------------------------------------------------------------
__global__ void __launch_bounds__(128)
head_kernel(const float* __restrict__ h1,
            const float* __restrict__ W1, const float* __restrict__ b1,
            const float* __restrict__ W2, const float* __restrict__ b2,
            const float* __restrict__ Wa, const float* __restrict__ ba,
            const float* __restrict__ mask,
            float* __restrict__ out) {
  __shared__ float h1s[128];
  __shared__ float hid[64];
  __shared__ float feat[128];
  __shared__ float lg[9];
  const int b = blockIdx.x;
  const int t = threadIdx.x;

  h1s[t] = h1[(size_t)b * 128 + t];
  __syncthreads();

  if (t < 64) {
    float s = b1[t];
    const float* w = W1 + (size_t)t * 128;
    #pragma unroll 8
    for (int k = 0; k < 128; ++k) s += h1s[k] * w[k];
    hid[t] = fmaxf(s, 0.0f);
  }
  __syncthreads();

  {
    float s = b2[t];
    const float* w = W2 + (size_t)t * 64;
    #pragma unroll 8
    for (int k = 0; k < 64; ++k) s += hid[k] * w[k];
    feat[t] = s;
  }
  __syncthreads();

  if (t < 9) {
    float s = ba[t];
    const float* w = Wa + (size_t)t * 128;
    #pragma unroll 8
    for (int k = 0; k < 128; ++k) s += feat[k] * w[k];
    lg[t] = s;
  }
  __syncthreads();

  if (t == 0) {
    float mx = lg[0];
    #pragma unroll
    for (int i = 1; i < 9; ++i) mx = fmaxf(mx, lg[i]);
    float ex[9];
    float sum = 0.0f;
    #pragma unroll
    for (int i = 0; i < 9; ++i) { ex[i] = __expf(lg[i] - mx); sum += ex[i]; }
    float inv = 1.0f / sum;
    const float* mrow = mask + (size_t)b * 9;
    float mk[9];
    float msum = 0.0f;
    #pragma unroll
    for (int i = 0; i < 9; ++i) { mk[i] = ex[i] * inv * mrow[i]; msum += mk[i]; }
    if (msum > 0.0f) {
      float minv = 1.0f / msum;
      #pragma unroll
      for (int i = 0; i < 9; ++i) out[(size_t)b * 9 + i] = mk[i] * minv;
    } else {
      #pragma unroll
      for (int i = 0; i < 9; ++i) out[(size_t)b * 9 + i] = ex[i] * inv;
    }
  }
}

// ---------------------------------------------------------------------------
extern "C" void kernel_launch(void* const* d_in, const int* in_sizes, int n_in,
                              void* d_out, int out_size, void* d_ws, size_t ws_size,
                              hipStream_t stream) {
  const int*   ability_ids = (const int*)  d_in[0];
  const int*   move_ids    = (const int*)  d_in[1];
  const float* numerical   = (const float*)d_in[2];
  const float* mask        = (const float*)d_in[3];
  const float* h0          = (const float*)d_in[4];
  const float* c0          = (const float*)d_in[5];
  const float* ability_emb = (const float*)d_in[6];
  const float* move_emb    = (const float*)d_in[7];
  const float* num_W       = (const float*)d_in[8];
  const float* num_b       = (const float*)d_in[9];
  const float* Wih         = (const float*)d_in[10];
  const float* Whh         = (const float*)d_in[11];
  const float* bih         = (const float*)d_in[12];
  const float* bhh         = (const float*)d_in[13];
  const float* W1          = (const float*)d_in[14];
  const float* b1          = (const float*)d_in[15];
  const float* W2          = (const float*)d_in[16];
  const float* b2          = (const float*)d_in[17];
  const float* Wa          = (const float*)d_in[18];
  const float* ba          = (const float*)d_in[19];
  float* out = (float*)d_out;

  // workspace layout (all offsets 256B aligned)
  char* ws = (char*)d_ws;
  _Float16* W_ext   = (_Float16*)(ws);                       // 512*7936*2  = 8,126,464
  _Float16* num_pre = (_Float16*)(ws + 8126464);             // 4096*128*2  = 1,048,576
  float*    gates   = (float*)   (ws + 8126464 + 1048576);   // 4096*512*4  = 8,388,608

  // 1) W_ext = f16([Wih | Whh])
  wext_kernel<<<(512 * KDIM) / 256, 256, 0, stream>>>(Wih, Whh, W_ext);
  // 2) numerical projection -> f16
  num_proj_kernel<<<(NB * 128) / 256, 256, 0, stream>>>(numerical, num_W, num_b, num_pre);
  // 3) WMMA GEMM with fused embedding gather (async B tile, double buffered)
  gates_gemm_kernel<<<(NB / 128) * (NGATE / 64), 256, 0, stream>>>(
      ability_ids, move_ids, ability_emb, move_emb, num_pre, h0, W_ext, gates);
  // 4) LSTM pointwise -> h1, c1 (into d_out)
  lstm_ew_kernel<<<(NB * 128) / 256, 256, 0, stream>>>(gates, bih, bhh, c0, out);
  // 5) MLP head + softmax + masked renorm (reads h1 from d_out)
  head_kernel<<<NB, 128, 0, stream>>>(out + NB * 9, W1, b1, W2, b2, Wa, ba, mask, out);
}
